// APLayer_65919158059653
// MI455X (gfx1250) — matrix operand order repaired and checked by reading, not compiled, not run on hardware
//
#include <hip/hip_runtime.h>
#include <hip/hip_bf16.h>
#include <hip/hip_fp16.h>

typedef __attribute__((ext_vector_type(16))) _Float16 v16h;
typedef __attribute__((ext_vector_type(8)))  float    v8f;

#define HDIM 64

// ---------------------------------------------------------------- zero scratch
__global__ void zero_ws_kernel(float* __restrict__ p, int n) {
    int i = blockIdx.x * blockDim.x + threadIdx.x;
    if (i < n) p[i] = 0.0f;
}

// ---------------------------------------------------------------- degrees
__global__ void degree_kernel(const int* __restrict__ src, const int* __restrict__ dst,
                              float* __restrict__ degS, float* __restrict__ degD, int E) {
    int e = blockIdx.x * blockDim.x + threadIdx.x;
    if (e < E) {
        atomicAdd(&degS[src[e]], 1.0f);
        atomicAdd(&degD[dst[e]], 1.0f);
    }
}

// ---------------------------------------------------------------- deg -> rsqrt(max(deg,1))
__global__ void norm_kernel(float* __restrict__ degS, float* __restrict__ degD, int n) {
    int i = blockIdx.x * blockDim.x + threadIdx.x;
    if (i < n) {
        degS[i] = rsqrtf(fmaxf(degS[i], 1.0f));
        degD[i] = rsqrtf(fmaxf(degD[i], 1.0f));
    }
}

// ---------------------------------------------------------------- edge scatter-add (SpMM)
// 16 threads per edge, each moves one float4 (16B) of the 64-float row.
__global__ void scatter_kernel(const float* __restrict__ feat,
                               const int* __restrict__ src, const int* __restrict__ dst,
                               const float* __restrict__ srcn,
                               float* __restrict__ agg, int E) {
    int gid = blockIdx.x * blockDim.x + threadIdx.x;
    int e = gid >> 4;
    int q = gid & 15;
    if (e < E) {
        int s = src[e];
        int d = dst[e];
        float sn = srcn[s];
        const float4* frow = (const float4*)(feat + (size_t)s * HDIM);
        float4 v = frow[q];
        float* arow = agg + (size_t)d * HDIM + q * 4;
        atomicAdd(arow + 0, v.x * sn);
        atomicAdd(arow + 1, v.y * sn);
        atomicAdd(arow + 2, v.z * sn);
        atomicAdd(arow + 3, v.w * sn);
    }
}

// ---------------------------------------------------------------- finalize (WMMA matvec + recurrence)
// One wave32 per 16 nodes. D = A(16x32 f16) x B(32x16 f16) + C(f32), twice (K=0..31, 32..63).
// B columns all hold halt_w chunk -> every column of D is the halting dot product.
__global__ void __launch_bounds__(32)
finalize_kernel(const float* __restrict__ feat, const float* __restrict__ agg,
                const float* __restrict__ dstn, const float* __restrict__ halt_w,
                const float* __restrict__ halt_b, const int* __restrict__ niter_p,
                float* __restrict__ xout, float* __restrict__ steps_out,
                float* __restrict__ rem_out, int n) {
    const int lane = threadIdx.x;
    const int base = blockIdx.x * 16;
    const int ni   = *niter_p;
    const float nif  = (float)ni;
    const float bias = halt_b[0];

    // ---- A matrix: lane l holds row base+(l&15), K chunk selected by lane half
    int node_a = base + (lane & 15);
    if (node_a >= n) node_a = n - 1;
    const int koff = (lane >> 4) * 8;
    const float dn_a = dstn[node_a];
    const float* rowp = agg + (size_t)node_a * HDIM;

    v16h a_lo, a_hi;
#pragma unroll
    for (int j = 0; j < 8; ++j) {
        a_lo[j]     = (_Float16)(rowp[koff + j]      * dn_a);  // K = koff..koff+7
        a_lo[8 + j] = (_Float16)(rowp[16 + koff + j] * dn_a);  // K = 16+koff..
        a_hi[j]     = (_Float16)(rowp[32 + koff + j] * dn_a);  // K = 32+koff..
        a_hi[8 + j] = (_Float16)(rowp[48 + koff + j] * dn_a);  // K = 48+koff..
    }

    // ---- B matrix: same K order as A (kb..kb+7, 16+kb..16+kb+7); broadcast weight to all 16 columns
    const int kb = (lane >> 4) * 8;
    v16h b_lo, b_hi;
#pragma unroll
    for (int j = 0; j < 16; ++j) {
        b_lo[j] = (_Float16)halt_w[kb + j + (j >= 8 ? 8 : 0)];        // K = kb+j (+8 for j>=8)
        b_hi[j] = (_Float16)halt_w[32 + kb + j + (j >= 8 ? 8 : 0)];   // K = 32+kb+j (+8 for j>=8)
    }

    v8f c = {};
    c = __builtin_amdgcn_wmma_f32_16x16x32_f16(false, a_lo, false, b_lo,
                                               (short)0, c, false, false);
    c = __builtin_amdgcn_wmma_f32_16x16x32_f16(false, a_hi, false, b_hi,
                                               (short)0, c, false, false);

    // D layout: VGPR r, lanes 0-15 -> row r; lanes 16-31 -> row r+8 (all columns equal).
    // Each lane runs the halting recurrence for its 8 rows.
    float stepsA[8], remA[8], sAA[8], sBA[8];
#pragma unroll
    for (int r = 0; r < 8; ++r) {
        float dot = c[r] + bias;
        float h = 1.0f / (1.0f + __expf(-dot));
        float steps = 1.0f, sumh = 0.0f, cA = 0.0f, cB = 0.0f;
        bool cont = true;
        for (int t = 0; t < ni; ++t) {
            float contf = cont ? 1.0f : 0.0f;
            bool  prob  = ((sumh + h) < 0.99f) && cont;
            float probf = prob ? 1.0f : 0.0f;
            steps += probf;
            sumh  += probf * h;
            bool  cond = prob && (steps < nif);
            float p    = cond ? sumh : (1.0f - sumh);
            if (t == 0) { cA += p * contf; cB += (1.0f - p) * contf; }  // old_prop = feature
            else        { cA += contf; }                                 // old_prop = prop: p+(1-p)=1
            cont = cont && prob;
        }
        float inv = 1.0f / steps;
        stepsA[r] = steps;
        remA[r]   = 1.0f - sumh;
        sAA[r]    = cA * inv;   // coefficient on prop (already /steps)
        sBA[r]    = cB * inv;   // coefficient on feature
    }

    // ---- write x rows (coalesced: 64 cols -> lane, lane+32) + per-node scalars
#pragma unroll
    for (int r = 0; r < 16; ++r) {
        const int srcl = (r < 8) ? 0 : 16;   // lane 0 holds rows 0..7, lane 16 rows 8..15
        const int rr   = r & 7;
        float sa = __shfl(sAA[rr],    srcl, 32);
        float sb = __shfl(sBA[rr],    srcl, 32);
        float st = __shfl(stepsA[rr], srcl, 32);
        float rm = __shfl(remA[rr],   srcl, 32);
        int node = base + r;
        if (node < n) {
            float dn = dstn[node];
            size_t off = (size_t)node * HDIM;
            float p0 = agg[off + lane]      * dn;
            float p1 = agg[off + lane + 32] * dn;
            xout[off + lane]      = sa * p0 + sb * feat[off + lane];
            xout[off + lane + 32] = sa * p1 + sb * feat[off + lane + 32];
            if (lane == 0) {
                steps_out[node] = st;
                rem_out[node]   = rm;
            }
        }
    }
}

// ---------------------------------------------------------------- launch
extern "C" void kernel_launch(void* const* d_in, const int* in_sizes, int n_in,
                              void* d_out, int out_size, void* d_ws, size_t ws_size,
                              hipStream_t stream) {
    const float* feat   = (const float*)d_in[0];
    const int*   src    = (const int*)  d_in[1];
    const int*   dst    = (const int*)  d_in[2];
    const float* halt_w = (const float*)d_in[3];
    const float* halt_b = (const float*)d_in[4];
    const int*   niter  = (const int*)  d_in[5];

    const int n = in_sizes[0] / HDIM;   // 50000
    const int E = in_sizes[1];          // 800000

    // workspace layout: [src_norm | dst_norm | agg(n x 64)]  (floats)
    float* degS = (float*)d_ws;
    float* degD = degS + n;
    float* agg  = degD + n;

    const int wsN = n * (HDIM + 2);
    zero_ws_kernel<<<(wsN + 255) / 256, 256, 0, stream>>>(degS, wsN);

    degree_kernel<<<(E + 255) / 256, 256, 0, stream>>>(src, dst, degS, degD, E);
    norm_kernel<<<(n + 255) / 256, 256, 0, stream>>>(degS, degD, n);

    const long long work = (long long)E * 16;
    scatter_kernel<<<(int)((work + 255) / 256), 256, 0, stream>>>(feat, src, dst, degS, agg, E);

    float* xout      = (float*)d_out;
    float* steps_out = xout + (size_t)n * HDIM;
    float* rem_out   = steps_out + n;
    finalize_kernel<<<(n + 15) / 16, 32, 0, stream>>>(feat, agg, degD, halt_w, halt_b,
                                                      niter, xout, steps_out, rem_out, n);
}
